// LerpChaining_60215441489998
// MI455X (gfx1250) — matrix operand order repaired and checked by reading, not compiled, not run on hardware
//
#include <hip/hip_runtime.h>
#include <math.h>

// Problem constants (match reference): B=4, W=32, N=2048, R=16
#define NB 4
#define NW 32
#define NN 2048
#define NR 16
#define NBW (NB * NW)          // 128 output rows
#define DB_SCALE 1024.0f       // lift db out of f16 subnormal range
#define INV_DB_SCALE (1.0f / 1024.0f)

typedef __attribute__((ext_vector_type(16))) _Float16 v16h;
typedef __attribute__((ext_vector_type(8)))  _Float16 v8h;
typedef __attribute__((ext_vector_type(4)))  _Float16 v4h;
typedef __attribute__((ext_vector_type(8)))  float    v8f;

union F16Frag { v16h v; v8h h[2]; };

// ---------------------------------------------------------------------------
// Phase A1: softmax of weights (32x32, row-wise) and equity_weight (32x2)
// ---------------------------------------------------------------------------
__global__ void prep_kernel(const float* __restrict__ wts,
                            const float* __restrict__ eq,
                            float* __restrict__ smW,     // [32 w][32 j]
                            float* __restrict__ ewv) {   // [32 w][2]
    int t = threadIdx.x;
    if (t < 32) {
        float buf[32];
        float m = -1e30f;
        #pragma unroll
        for (int j = 0; j < 32; ++j) { buf[j] = wts[t * 32 + j]; m = fmaxf(m, buf[j]); }
        float s = 0.0f;
        #pragma unroll
        for (int j = 0; j < 32; ++j) { buf[j] = expf(buf[j] - m); s += buf[j]; }
        float inv = 1.0f / s;
        #pragma unroll
        for (int j = 0; j < 32; ++j) smW[t * 32 + j] = buf[j] * inv;
    } else if (t < 64) {
        int w = t - 32;
        float a = eq[w * 2 + 0], b = eq[w * 2 + 1];
        float m = fmaxf(a, b);
        float ea = expf(a - m), eb = expf(b - m);
        float inv = 1.0f / (ea + eb);
        ewv[w * 2 + 0] = ea * inv;
        ewv[w * 2 + 1] = eb * inv;
    }
}

// ---------------------------------------------------------------------------
// Phase A2: convert inputs (128 x 2048 fp32) -> f16, and zero the accumulator.
// 65536 threads, one float4 each.
// ---------------------------------------------------------------------------
__global__ void cvt_kernel(const float* __restrict__ x,
                           _Float16* __restrict__ x16,
                           float* __restrict__ outAcc) {
    int i = blockIdx.x * blockDim.x + threadIdx.x;   // float4 index, 0..65535
    float4 d = ((const float4*)x)[i];
    v4h h = { (_Float16)d.x, (_Float16)d.y, (_Float16)d.z, (_Float16)d.w };
    ((v4h*)x16)[i] = h;
    float4 z = { 0.0f, 0.0f, 0.0f, 0.0f };
    ((float4*)outAcc)[i] = z;
}

// ---------------------------------------------------------------------------
// Phase B: for each (m-tile, r): C128x64 (fwd) = X @ db[r], C (bwd) = X @ db[r]^T
// over full K=2048; epilogue applies softmax weights and atomically accumulates.
// Block: 256 threads = 8 waves. Wave (wr, wcg) owns rows [wr*32,+32) cols [wcg*32,+32).
// Double-buffered LDS: one barrier per k-step; next tile's global loads are
// issued right after the barrier so they fly underneath the WMMAs.
// ---------------------------------------------------------------------------
#define LDSP 40               // LDS pitch in halves: 80 B rows -> frag reads 16B aligned
#define BUFH (64 * LDSP)      // halves per buffer

struct Stage { float4 b[2]; float4 f[2]; };

__device__ __forceinline__ void load_stage(const float* __restrict__ db,
                                           int m0, int kk, int t, Stage& s) {
    #pragma unroll
    for (int pass = 0; pass < 2; ++pass) {
        const int f = t + pass * 256;            // 0..511 (512 float4 per tile)
        const int row  = f >> 3, c4  = f & 7;    // bwd tile coords
        s.b[pass] = *(const float4*)(db + (size_t)(m0 + row) * NN + kk + c4 * 4);
        const int grow = f >> 4, gc4 = f & 15;   // fwd tile coords
        s.f[pass] = *(const float4*)(db + (size_t)(kk + grow) * NN + m0 + gc4 * 4);
    }
}

__device__ __forceinline__ void store_stage(_Float16* __restrict__ ldsB,
                                            _Float16* __restrict__ ldsF,
                                            int t, const Stage& s) {
    #pragma unroll
    for (int pass = 0; pass < 2; ++pass) {
        const int f = t + pass * 256;
        {   // bwd tile: straight copy, 8-byte f16 stores
            const int row = f >> 3, c4 = f & 7;
            const float4 d = s.b[pass];
            v4h h = { (_Float16)(d.x * DB_SCALE), (_Float16)(d.y * DB_SCALE),
                      (_Float16)(d.z * DB_SCALE), (_Float16)(d.w * DB_SCALE) };
            *(v4h*)(ldsB + row * LDSP + c4 * 4) = h;
        }
        {   // fwd tile: transposed scatter of 4 halves
            const int grow = f >> 4, gc4 = f & 15;
            const float4 d = s.f[pass];
            ldsF[(gc4 * 4 + 0) * LDSP + grow] = (_Float16)(d.x * DB_SCALE);
            ldsF[(gc4 * 4 + 1) * LDSP + grow] = (_Float16)(d.y * DB_SCALE);
            ldsF[(gc4 * 4 + 2) * LDSP + grow] = (_Float16)(d.z * DB_SCALE);
            ldsF[(gc4 * 4 + 3) * LDSP + grow] = (_Float16)(d.w * DB_SCALE);
        }
    }
}

__global__ __launch_bounds__(256) void gemm_kernel(
        const _Float16* __restrict__ X16,     // [128][2048] f16
        const float* __restrict__ db_all,     // [16][2048][2048] fp32
        const float* __restrict__ smW,        // [32 w][32 j]
        float* __restrict__ outAcc) {         // [128][2048] fp32
    const int m0 = blockIdx.x * 64;
    const int r  = blockIdx.y;
    const float* __restrict__ db = db_all + (size_t)r * NN * NN;

    __shared__ _Float16 ldsF[2 * BUFH];   // fwd tile, stored transposed: [m][k]
    __shared__ _Float16 ldsB[2 * BUFH];   // bwd tile, row-major copy:    [m][k]
    __shared__ float wfS[32];
    __shared__ float wbS[32];

    const int t = threadIdx.x;
    if (t < 32) {
        wfS[t] = smW[t * 32 + r]      * INV_DB_SCALE;
        wbS[t] = smW[t * 32 + 16 + r] * INV_DB_SCALE;
    }

    const int wave = t >> 5;
    const int lane = t & 31;
    const int wr   = wave & 3;     // row group: rows [wr*32, wr*32+32)
    const int wcg  = wave >> 2;    // col group: cols [wcg*32, wcg*32+32)
    const int mloc = lane & 15;
    const int kh   = lane >> 4;    // which K-half of the fragment this lane holds

    v8f accF[2][2] = {};           // [row-subtile][col-subtile]
    v8f accB[2][2] = {};

    Stage st;
    load_stage(db, m0, 0, t, st);

    int cur = 0;
    #pragma unroll 2
    for (int kk = 0; kk < NN; kk += 32) {
        // store the prefetched tiles into the current buffer
        store_stage(ldsB + cur * BUFH, ldsF + cur * BUFH, t, st);
        __syncthreads();

        // immediately issue next tile's global loads (in flight under WMMAs)
        if (kk + 32 < NN) load_stage(db, m0, kk + 32, t, st);

        // ---- A fragments straight from global (L2-resident X16) ----
        // A frag in contiguous k order (B matches): lane m (kh half) holds K = kh*16+h (h<16)
        F16Frag a0, a1;
        {
            const _Float16* ap = X16 + (size_t)(wr * 32 + mloc) * NN + kk + kh * 16;
            a0.h[0] = *(const v8h*)(ap);
            a0.h[1] = *(const v8h*)(ap + 8);
            const _Float16* ap1 = ap + (size_t)16 * NN;
            a1.h[0] = *(const v8h*)(ap1);
            a1.h[1] = *(const v8h*)(ap1 + 8);
        }

        // ---- B fragments from LDS + 8 WMMAs ----
        const _Float16* bufF = ldsF + cur * BUFH;
        const _Float16* bufB = ldsB + cur * BUFH;
        #pragma unroll
        for (int cb = 0; cb < 2; ++cb) {
            const int nrow = wcg * 32 + cb * 16 + mloc;
            F16Frag bF, bB;
            const _Float16* pf = bufF + nrow * LDSP + kh * 16;
            bF.h[0] = *(const v8h*)(pf);
            bF.h[1] = *(const v8h*)(pf + 8);
            const _Float16* pb = bufB + nrow * LDSP + kh * 16;
            bB.h[0] = *(const v8h*)(pb);
            bB.h[1] = *(const v8h*)(pb + 8);

            accF[0][cb] = __builtin_amdgcn_wmma_f32_16x16x32_f16(
                false, a0.v, false, bF.v, (short)0, accF[0][cb], false, false);
            accF[1][cb] = __builtin_amdgcn_wmma_f32_16x16x32_f16(
                false, a1.v, false, bF.v, (short)0, accF[1][cb], false, false);
            accB[0][cb] = __builtin_amdgcn_wmma_f32_16x16x32_f16(
                false, a0.v, false, bB.v, (short)0, accB[0][cb], false, false);
            accB[1][cb] = __builtin_amdgcn_wmma_f32_16x16x32_f16(
                false, a1.v, false, bB.v, (short)0, accB[1][cb], false, false);
        }
        cur ^= 1;
    }

    // ---- epilogue: weight by softmax(r,w) and accumulate ----
    // C/D layout: VGPR j holds (M = j + 8*kh, N = lane&15)
    #pragma unroll
    for (int rb = 0; rb < 2; ++rb) {
        #pragma unroll
        for (int cb = 0; cb < 2; ++cb) {
            const int col = m0 + wcg * 32 + cb * 16 + mloc;
            #pragma unroll
            for (int j = 0; j < 8; ++j) {
                const int row = wr * 32 + rb * 16 + kh * 8 + j;
                const int w = row & 31;
                const float v = wfS[w] * accF[rb][cb][j] + wbS[w] * accB[rb][cb][j];
                atomicAdd(outAcc + (size_t)row * NN + col, v);
            }
        }
    }
}

// ---------------------------------------------------------------------------
// Phase C: out = (1 - exp(-acc)) * ew0[w] + inputs * ew1[w]
// ---------------------------------------------------------------------------
__global__ void final_kernel(const float* __restrict__ outAcc,
                             const float* __restrict__ x,
                             const float* __restrict__ ewv,
                             float* __restrict__ out) {
    int i = blockIdx.x * blockDim.x + threadIdx.x;
    if (i >= NBW * NN) return;
    int w = (i >> 11) & 31;                 // row = i / 2048; w = row % 32
    float a = outAcc[i];
    float v = 1.0f - expf(-a);
    out[i] = v * ewv[w * 2 + 0] + x[i] * ewv[w * 2 + 1];
}

// ---------------------------------------------------------------------------
extern "C" void kernel_launch(void* const* d_in, const int* in_sizes, int n_in,
                              void* d_out, int out_size, void* d_ws, size_t ws_size,
                              hipStream_t stream) {
    const float* inputs   = (const float*)d_in[0];   // (4, 32, 2048)
    const float* database = (const float*)d_in[1];   // (16, 2048, 2048)
    const float* weights  = (const float*)d_in[2];   // (32, 32)
    const float* equity   = (const float*)d_in[3];   // (32, 2)
    float* out = (float*)d_out;

    char* ws = (char*)d_ws;
    _Float16* X16   = (_Float16*)ws;                                   // 512 KB
    float*    acc   = (float*)(ws + (size_t)512 * 1024);               // 1 MB
    float*    smW   = (float*)(ws + (size_t)512 * 1024 + 1024 * 1024); // 4 KB
    float*    ewv   = smW + 32 * 32;                                   // 256 B

    prep_kernel<<<1, 64, 0, stream>>>(weights, equity, smW, ewv);
    cvt_kernel<<<256, 256, 0, stream>>>(inputs, X16, acc);
    gemm_kernel<<<dim3(NN / 64, NR), 256, 0, stream>>>(X16, database, smW, acc);
    final_kernel<<<(NBW * NN + 255) / 256, 256, 0, stream>>>(acc, inputs, ewv, out);
}
